// HGTEncoder_87411174409065
// MI455X (gfx1250) — compile-verified, matrix operand order checked by reading, not run on hardware
//
#include <hip/hip_runtime.h>
#include <hip/hip_fp16.h>
#include <math.h>

typedef __attribute__((ext_vector_type(16))) _Float16 v16h;
typedef __attribute__((ext_vector_type(8)))  _Float16 v8h;
typedef __attribute__((ext_vector_type(8)))  float    v8f;

constexpr int NH  = 8;     // heads
constexpr int HD  = 16;    // dim per head
constexpr int NF  = 128;   // features
constexpr int NF3 = 384;   // 3*F

__device__ __forceinline__ void atomicMaxF(float* addr, float val) {
  if (val >= 0.0f) atomicMax((int*)addr, __float_as_int(val));
  else             atomicMin((unsigned int*)addr, (unsigned int)__float_as_int(val));
}

// ---------------------------------------------------------------- utilities
__global__ void fill4_kernel(float4* __restrict__ p, float v, int n4) {
  int i = blockIdx.x * blockDim.x + threadIdx.x;
  if (i < n4) p[i] = make_float4(v, v, v, v);
}
__global__ void copy4_kernel(const float4* __restrict__ s, float4* __restrict__ d, int n4) {
  int i = blockIdx.x * blockDim.x + threadIdx.x;
  if (i < n4) d[i] = s[i];
}
__global__ void f32_to_f16_kernel(const float* __restrict__ s, _Float16* __restrict__ d, int n) {
  int i = blockIdx.x * blockDim.x + threadIdx.x;
  if (i < n) d[i] = (_Float16)s[i];
}

// Pre-swizzle Ak/Av [LR][H][16][16] f32 -> per-lane fragment order:
// dst[lr][h][lane][i] = src[lr][h][d = (lane>>4)*8 + i][e = lane&15],  i in 0..7
__global__ void swizzle_head_mats_kernel(const float* __restrict__ src, _Float16* __restrict__ dst,
                                         int n /* = LR*H*32*8 */) {
  int idx = blockIdx.x * blockDim.x + threadIdx.x;
  if (idx >= n) return;
  int i    = idx & 7;
  int lane = (idx >> 3) & 31;
  int lrh  = idx >> 8;           // (lr*H + h)
  int d = (lane >> 4) * 8 + i;
  int e = lane & 15;
  dst[idx] = (_Float16)src[(size_t)(lrh * HD + d) * HD + e];
}

// ------------------------------------------------- KQV projection (WMMA GEMM)
// C[M x 384] = X[M x 128] @ W + b ; block = 64 rows x 64 cols, 4 waves
__global__ __launch_bounds__(128)
void gemm_kqv_kernel(const float* __restrict__ X, const _Float16* __restrict__ W16,
                     const float* __restrict__ bias,
                     float* __restrict__ Kb, float* __restrict__ Qb, float* __restrict__ Vb,
                     int M, int node_off)
{
  __shared__ alignas(16) _Float16 As[64][136];        // 64 x 128 (+8 pad)
  __shared__ alignas(16) _Float16 Bs[4][4][32][16];   // [wave][kk][lane][elem]
  const int tid = threadIdx.x;
  const int r0  = blockIdx.x * 64;
  const int c0  = blockIdx.y * 64;
  const bool full = (r0 + 64 <= M);

  if (full) {
    for (int i = tid; i < 64 * 128; i += 128) {
      int r = i >> 7, c = i & 127;
      As[r][c] = (_Float16)X[(size_t)(r0 + r) * NF + c];
    }
  } else {
    for (int i = tid; i < 64 * 128; i += 128) {
      int r = i >> 7, c = i & 127;
      int row = r0 + r;
      As[r][c] = (row < M) ? (_Float16)X[(size_t)row * NF + c] : (_Float16)0.0f;
    }
  }
  // stage B pre-swizzled into fragment order (global reads coalesced over col)
  for (int idx = tid; idx < 128 * 64; idx += 128) {
    int col = idx & 63;
    int kr  = idx >> 6;                 // 0..127 = global K row
    int kk  = kr >> 5;
    int k5  = kr & 31;
    int hf  = (k5 >> 3) & 1;
    int el  = (k5 & 7) + ((k5 >> 4) << 3);
    int ln  = hf * 16 + (col & 15);
    Bs[col >> 4][kk][ln][el] = W16[(size_t)kr * NF3 + c0 + col];
  }
  __syncthreads();

  const int lane = tid & 31, wv = tid >> 5;
  const int mr = lane & 15, half = lane >> 4;

  v16h bfrag[4];
#pragma unroll
  for (int kk = 0; kk < 4; ++kk) {
    const v8h* pb = (const v8h*)&Bs[wv][kk][lane][0];
    v8h b0 = pb[0], b1 = pb[1];
#pragma unroll
    for (int i = 0; i < 8; ++i) { bfrag[kk][i] = b0[i]; bfrag[kk][i + 8] = b1[i]; }
  }

  v8f accs[4];
#pragma unroll
  for (int rt = 0; rt < 4; ++rt) {
    v8f acc = {};
#pragma unroll
    for (int kk = 0; kk < 4; ++kk) {
      const v8h* pa0 = (const v8h*)&As[rt * 16 + mr][kk * 32 + half * 8];
      const v8h* pa1 = (const v8h*)&As[rt * 16 + mr][kk * 32 + 16 + half * 8];
      v8h a0 = pa0[0], a1 = pa1[0];
      v16h a;
#pragma unroll
      for (int i = 0; i < 8; ++i) { a[i] = a0[i]; a[i + 8] = a1[i]; }
      acc = __builtin_amdgcn_wmma_f32_16x16x32_f16(false, a, false, bfrag[kk], (short)0, acc, false, false);
    }
    accs[rt] = acc;
  }

  // wave-uniform K/Q/V segment routing (a wave's 16 cols never cross a segment)
  const int cbase = c0 + wv * 16;            // uniform per wave
  const int cseg  = cbase >> 7;              // 0=K, 1=Q, 2=V
  const int ccol  = (cbase & 127) + mr;      // column within segment
  float* __restrict__ dstP = (cseg == 0) ? Kb : (cseg == 1) ? Qb : Vb;
  const float bval = bias[cbase + mr];       // hoisted, loaded once
  float* __restrict__ outp = dstP + (size_t)node_off * NF + ccol;

  if (full) {
#pragma unroll
    for (int rt = 0; rt < 4; ++rt)
#pragma unroll
      for (int j = 0; j < 8; ++j) {
        int r = r0 + rt * 16 + j + 8 * half;
        outp[(size_t)r * NF] = accs[rt][j] + bval;
      }
  } else {
#pragma unroll
    for (int rt = 0; rt < 4; ++rt)
#pragma unroll
      for (int j = 0; j < 8; ++j) {
        int r = r0 + rt * 16 + j + 8 * half;
        if (r < M) outp[(size_t)r * NF] = accs[rt][j] + bval;
      }
  }
}

// ------------------------------------- per-relation per-head 16x16 transforms
// k_r[n,h,:] = k[n,h,:] @ Ak[h]  (K=16 -> only fragment elems 0..7 live)
__global__ __launch_bounds__(128)
void head_transform_kernel(const float* __restrict__ Ksrc, const float* __restrict__ Vsrc,
                           const _Float16* __restrict__ AkS, const _Float16* __restrict__ AvS,
                           float* __restrict__ KR, float* __restrict__ VR, int N)
{
  const int tid = threadIdx.x, lane = tid & 31, wv = tid >> 5;
  const int h    = blockIdx.y * 4 + wv;
  const int tile = blockIdx.x;
  const int mr = lane & 15, half = lane >> 4;
  const int row = tile * 16 + mr;

  v16h aK = {}, aV = {}, bK = {}, bV = {};
  if (row < N) {
    // elems 0..7 map to k = half*8 + i : one contiguous 8-float run
    const float* kp = Ksrc + (size_t)row * NF + h * HD + half * 8;
    const float* vp = Vsrc + (size_t)row * NF + h * HD + half * 8;
#pragma unroll
    for (int i = 0; i < 8; ++i) {
      aK[i] = (_Float16)kp[i];
      aV[i] = (_Float16)vp[i];
    }
  }
  {
    const v8h* pk = (const v8h*)(AkS + ((size_t)h * 32 + lane) * 8);
    const v8h* pv = (const v8h*)(AvS + ((size_t)h * 32 + lane) * 8);
    v8h bk = *pk, bv = *pv;
#pragma unroll
    for (int i = 0; i < 8; ++i) { bK[i] = bk[i]; bV[i] = bv[i]; }
  }
  v8f accK = {}, accV = {};
  accK = __builtin_amdgcn_wmma_f32_16x16x32_f16(false, aK, false, bK, (short)0, accK, false, false);
  accV = __builtin_amdgcn_wmma_f32_16x16x32_f16(false, aV, false, bV, (short)0, accV, false, false);
#pragma unroll
  for (int j = 0; j < 8; ++j) {
    int r = tile * 16 + j + 8 * half;
    if (r < N) {
      KR[(size_t)r * NF + h * HD + mr] = accK[j];
      VR[(size_t)r * NF + h * HD + mr] = accV[j];
    }
  }
}

// ------------------------------------------------------------ edge passes
__global__ void edge_logits_kernel(const int* __restrict__ src, const int* __restrict__ dst, int E,
                                   const float* __restrict__ Qb, const float* __restrict__ KR,
                                   const float* __restrict__ prel_h, int dst_off,
                                   float* __restrict__ alphaB, float* __restrict__ amax)
{
  int idx = blockIdx.x * blockDim.x + threadIdx.x;
  if (idx >= E * NH) return;
  int e = idx >> 3, h = idx & 7;
  int s = src[e];
  int dg = dst[e] + dst_off;
  const float4* q = (const float4*)(Qb + (size_t)dg * NF + h * HD);
  const float4* k = (const float4*)(KR + (size_t)s  * NF + h * HD);
  float a = 0.f;
#pragma unroll
  for (int j = 0; j < 4; ++j) {
    float4 qv = q[j], kv = k[j];
    a += qv.x * kv.x + qv.y * kv.y + qv.z * kv.z + qv.w * kv.w;
  }
  a *= prel_h[h] * 0.25f;   // 1/sqrt(16)
  alphaB[(size_t)e * NH + h] = a;
  atomicMaxF(amax + (size_t)dg * NH + h, a);
}

__global__ void edge_expsum_kernel(const int* __restrict__ dst, int E, int dst_off,
                                   float* __restrict__ alphaB, const float* __restrict__ amax,
                                   float* __restrict__ denom)
{
  int idx = blockIdx.x * blockDim.x + threadIdx.x;
  if (idx >= E * NH) return;
  int e = idx >> 3, h = idx & 7;
  int dg = dst[e] + dst_off;
  float ex = expf(alphaB[(size_t)e * NH + h] - amax[(size_t)dg * NH + h]);
  alphaB[(size_t)e * NH + h] = ex;
  atomicAdd(denom + (size_t)dg * NH + h, ex);
}

__global__ __launch_bounds__(128)
void edge_scatter_kernel(const int* __restrict__ src, const int* __restrict__ dst, int dst_off,
                         const float* __restrict__ alphaB, const float* __restrict__ denom,
                         const float* __restrict__ VR, float* __restrict__ agg)
{
  int e = blockIdx.x, f = threadIdx.x, h = f >> 4;
  int s = src[e];
  int dg = dst[e] + dst_off;
  float w = alphaB[(size_t)e * NH + h] / denom[(size_t)dg * NH + h];
  atomicAdd(agg + (size_t)dg * NF + f, w * VR[(size_t)s * NF + f]);
}

// --------------------------- output projection + skip gate + relu residual
__global__ __launch_bounds__(128)
void gemm_out_kernel(const float* __restrict__ AGG, const _Float16* __restrict__ W16,
                     const float* __restrict__ bias, const float* __restrict__ skip_p,
                     float* __restrict__ Xio, int M)
{
  __shared__ alignas(16) _Float16 As[64][136];
  __shared__ alignas(16) _Float16 Bs[4][4][32][16];
  const int tid = threadIdx.x;
  const int r0  = blockIdx.x * 64;
  const int c0  = blockIdx.y * 64;
  const bool full = (r0 + 64 <= M);

  if (full) {
    for (int i = tid; i < 64 * 128; i += 128) {
      int r = i >> 7, c = i & 127;
      float v = AGG[(size_t)(r0 + r) * NF + c];
      float g = 0.5f * v * (1.0f + erff(v * 0.70710678118654752f));
      As[r][c] = (_Float16)g;
    }
  } else {
    for (int i = tid; i < 64 * 128; i += 128) {
      int r = i >> 7, c = i & 127;
      int row = r0 + r;
      float v = (row < M) ? AGG[(size_t)row * NF + c] : 0.0f;
      float g = 0.5f * v * (1.0f + erff(v * 0.70710678118654752f));
      As[r][c] = (_Float16)g;
    }
  }
  for (int idx = tid; idx < 128 * 64; idx += 128) {
    int col = idx & 63;
    int kr  = idx >> 6;
    int kk  = kr >> 5;
    int k5  = kr & 31;
    int hf  = (k5 >> 3) & 1;
    int el  = (k5 & 7) + ((k5 >> 4) << 3);
    int ln  = hf * 16 + (col & 15);
    Bs[col >> 4][kk][ln][el] = W16[(size_t)kr * NF + c0 + col];
  }
  __syncthreads();

  const int lane = tid & 31, wv = tid >> 5;
  const int mr = lane & 15, half = lane >> 4;

  v16h bfrag[4];
#pragma unroll
  for (int kk = 0; kk < 4; ++kk) {
    const v8h* pb = (const v8h*)&Bs[wv][kk][lane][0];
    v8h b0 = pb[0], b1 = pb[1];
#pragma unroll
    for (int i = 0; i < 8; ++i) { bfrag[kk][i] = b0[i]; bfrag[kk][i + 8] = b1[i]; }
  }

  v8f accs[4];
#pragma unroll
  for (int rt = 0; rt < 4; ++rt) {
    v8f acc = {};
#pragma unroll
    for (int kk = 0; kk < 4; ++kk) {
      const v8h* pa0 = (const v8h*)&As[rt * 16 + mr][kk * 32 + half * 8];
      const v8h* pa1 = (const v8h*)&As[rt * 16 + mr][kk * 32 + 16 + half * 8];
      v8h a0 = pa0[0], a1 = pa1[0];
      v16h a;
#pragma unroll
      for (int i = 0; i < 8; ++i) { a[i] = a0[i]; a[i + 8] = a1[i]; }
      acc = __builtin_amdgcn_wmma_f32_16x16x32_f16(false, a, false, bfrag[kk], (short)0, acc, false, false);
    }
    accs[rt] = acc;
  }

  const float a_s  = 1.0f / (1.0f + expf(-skip_p[0]));
  const int   c    = c0 + wv * 16 + mr;
  const float bval = bias[c];                 // hoisted, loaded once
  float* __restrict__ xp = Xio + c;

  if (full) {
#pragma unroll
    for (int rt = 0; rt < 4; ++rt)
#pragma unroll
      for (int j = 0; j < 8; ++j) {
        int r = r0 + rt * 16 + j + 8 * half;
        float o  = accs[rt][j] + bval;
        float xo = xp[(size_t)r * NF];
        float nv = a_s * o + (1.0f - a_s) * xo;
        xp[(size_t)r * NF] = fmaxf(nv, 0.0f) + xo;
      }
  } else {
#pragma unroll
    for (int rt = 0; rt < 4; ++rt)
#pragma unroll
      for (int j = 0; j < 8; ++j) {
        int r = r0 + rt * 16 + j + 8 * half;
        if (r < M) {
          float o  = accs[rt][j] + bval;
          float xo = xp[(size_t)r * NF];
          float nv = a_s * o + (1.0f - a_s) * xo;
          xp[(size_t)r * NF] = fmaxf(nv, 0.0f) + xo;
        }
      }
  }
}

// ------------------------------------------------------------------ driver
extern "C" void kernel_launch(void* const* d_in, const int* in_sizes, int n_in,
                              void* d_out, int out_size, void* d_ws, size_t ws_size,
                              hipStream_t stream) {
  const float* xg   = (const float*)d_in[0];
  const float* xd   = (const float*)d_in[1];
  const float* xr   = (const float*)d_in[2];
  const float* Wkqv = (const float*)d_in[3];   // [2][3][128][384]
  const float* bkqv = (const float*)d_in[4];   // [2][3][384]
  const float* Ak   = (const float*)d_in[5];   // [2][4][8][16][16]
  const float* Av   = (const float*)d_in[6];
  const float* prel = (const float*)d_in[7];   // [2][4][8]
  const float* Wout = (const float*)d_in[8];   // [2][3][128][128]
  const float* bout = (const float*)d_in[9];   // [2][3][128]
  const float* skip = (const float*)d_in[10];  // [2][3]
  const int*   ei[4] = {(const int*)d_in[11], (const int*)d_in[12],
                        (const int*)d_in[13], (const int*)d_in[14]};

  static const int NS_[3]   = {50000, 20000, 10000};
  static const int OFFS_[4] = {0, 50000, 70000, 80000};
  static const int ES_[4]   = {300000, 150000, 100000, 50000};
  static const int EOFF_[4] = {0, 300000, 450000, 550000};
  static const int KVOFF_[4]= {0, 50000, 100000, 110000};
  static const int ST_[4]   = {0, 0, 2, 2};
  static const int DT_[4]   = {0, 1, 0, 1};
  const int NTOT = 80000;
  const int KVROWS = 120000;

  char* ws = (char*)d_ws;
  size_t off = 0;
  auto alloc = [&](size_t bytes) -> void* {
    void* p = ws + off;
    off = (off + bytes + 255) & ~(size_t)255;
    return p;
  };
  _Float16* Wkqv16 = (_Float16*)alloc((size_t)2 * 3 * NF * NF3 * sizeof(_Float16));
  _Float16* Wout16 = (_Float16*)alloc((size_t)2 * 3 * NF * NF  * sizeof(_Float16));
  _Float16* AkS    = (_Float16*)alloc((size_t)2 * 4 * NH * 32 * 8 * sizeof(_Float16));
  _Float16* AvS    = (_Float16*)alloc((size_t)2 * 4 * NH * 32 * 8 * sizeof(_Float16));
  float* Kb     = (float*)alloc((size_t)NTOT * NF * sizeof(float));
  float* Qb     = (float*)alloc((size_t)NTOT * NF * sizeof(float));
  float* Vb     = (float*)alloc((size_t)NTOT * NF * sizeof(float));
  float* KR     = (float*)alloc((size_t)KVROWS * NF * sizeof(float));
  float* VR     = (float*)alloc((size_t)KVROWS * NF * sizeof(float));
  float* alphaB = (float*)alloc((size_t)600000 * NH * sizeof(float));
  float* amax   = (float*)alloc((size_t)NTOT * NH * sizeof(float));
  float* denom  = (float*)alloc((size_t)NTOT * NH * sizeof(float));
  float* agg    = (float*)alloc((size_t)NTOT * NF * sizeof(float));

  // weight conversions / swizzles (deterministic, every call)
  {
    int n1 = 2 * 3 * NF * NF3;
    f32_to_f16_kernel<<<(n1 + 255) / 256, 256, 0, stream>>>(Wkqv, Wkqv16, n1);
    int n2 = 2 * 3 * NF * NF;
    f32_to_f16_kernel<<<(n2 + 255) / 256, 256, 0, stream>>>(Wout, Wout16, n2);
    int n3 = 2 * 4 * NH * 32 * 8;
    swizzle_head_mats_kernel<<<(n3 + 255) / 256, 256, 0, stream>>>(Ak, AkS, n3);
    swizzle_head_mats_kernel<<<(n3 + 255) / 256, 256, 0, stream>>>(Av, AvS, n3);
  }

  // x lives in d_out (concat order == output order), updated in place
  float* X = (float*)d_out;
  {
    int n0 = NS_[0] * NF / 4, n1 = NS_[1] * NF / 4, n2 = NS_[2] * NF / 4;
    copy4_kernel<<<(n0 + 255) / 256, 256, 0, stream>>>((const float4*)xg, (float4*)(X + (size_t)OFFS_[0] * NF), n0);
    copy4_kernel<<<(n1 + 255) / 256, 256, 0, stream>>>((const float4*)xd, (float4*)(X + (size_t)OFFS_[1] * NF), n1);
    copy4_kernel<<<(n2 + 255) / 256, 256, 0, stream>>>((const float4*)xr, (float4*)(X + (size_t)OFFS_[2] * NF), n2);
  }

  for (int l = 0; l < 2; ++l) {
    // 1) KQV projections
    for (int t = 0; t < 3; ++t) {
      dim3 g((NS_[t] + 63) / 64, NF3 / 64);
      gemm_kqv_kernel<<<g, 128, 0, stream>>>(
          X + (size_t)OFFS_[t] * NF,
          Wkqv16 + (size_t)(l * 3 + t) * NF * NF3,
          bkqv + (size_t)(l * 3 + t) * NF3,
          Kb, Qb, Vb, NS_[t], OFFS_[t]);
    }
    // 2) per-relation head transforms
    for (int r = 0; r < 4; ++r) {
      int st = ST_[r], N = NS_[st];
      dim3 g((N + 15) / 16, 2);
      head_transform_kernel<<<g, 128, 0, stream>>>(
          Kb + (size_t)OFFS_[st] * NF, Vb + (size_t)OFFS_[st] * NF,
          AkS + (size_t)(l * 4 + r) * NH * 32 * 8,
          AvS + (size_t)(l * 4 + r) * NH * 32 * 8,
          KR + (size_t)KVOFF_[r] * NF, VR + (size_t)KVOFF_[r] * NF, N);
    }
    // 3) init softmax accumulators
    {
      int nh4 = NTOT * NH / 4, nf4 = NTOT * NF / 4;
      fill4_kernel<<<(nh4 + 255) / 256, 256, 0, stream>>>((float4*)amax, -INFINITY, nh4);
      fill4_kernel<<<(nh4 + 255) / 256, 256, 0, stream>>>((float4*)denom, 0.0f, nh4);
      fill4_kernel<<<(nf4 + 255) / 256, 256, 0, stream>>>((float4*)agg, 0.0f, nf4);
    }
    // 4) edge logits + segment max
    for (int r = 0; r < 4; ++r) {
      int E = ES_[r];
      edge_logits_kernel<<<(E * NH + 255) / 256, 256, 0, stream>>>(
          ei[r], ei[r] + E, E, Qb,
          KR + (size_t)KVOFF_[r] * NF,
          prel + (size_t)(l * 4 + r) * NH, OFFS_[DT_[r]],
          alphaB + (size_t)EOFF_[r] * NH, amax);
    }
    // 5) exp + segment sum
    for (int r = 0; r < 4; ++r) {
      int E = ES_[r];
      edge_expsum_kernel<<<(E * NH + 255) / 256, 256, 0, stream>>>(
          ei[r] + E, E, OFFS_[DT_[r]],
          alphaB + (size_t)EOFF_[r] * NH, amax, denom);
    }
    // 6) weighted scatter of v
    for (int r = 0; r < 4; ++r) {
      int E = ES_[r];
      edge_scatter_kernel<<<E, 128, 0, stream>>>(
          ei[r], ei[r] + E, OFFS_[DT_[r]],
          alphaB + (size_t)EOFF_[r] * NH, denom,
          VR + (size_t)KVOFF_[r] * NF, agg);
    }
    // 7) output projection + gate + relu residual (in-place on X)
    for (int t = 0; t < 3; ++t) {
      dim3 g((NS_[t] + 63) / 64, 2);
      gemm_out_kernel<<<g, 128, 0, stream>>>(
          agg + (size_t)OFFS_[t] * NF,
          Wout16 + (size_t)(l * 3 + t) * NF * NF,
          bout + (size_t)(l * 3 + t) * NF,
          skip + (size_t)(l * 3 + t),
          X + (size_t)OFFS_[t] * NF, NS_[t]);
    }
  }
}
